// Faceformer_32847909880501
// MI455X (gfx1250) — matrix operand order repaired and checked by reading, not compiled, not run on hardware
//
#include <hip/hip_runtime.h>
#include <math.h>

#define TT 120
#define TP 128                  // padded row count (multiple of 16)
#define DD 64
#define NEGV -1000000000.0f
#define SCALEV 0.17677669529663687f  // 1/sqrt(32)

typedef __attribute__((ext_vector_type(16))) _Float16 v16h;
typedef __attribute__((ext_vector_type(8)))  float    v8f;

struct Par {
  const float *audio,*W_audio,*b_audio,*obj_W;
  const float *Wq,*bq,*Wk,*bk,*Wv,*bv,*Wo,*bo;
  const float *Wv2,*bv2,*Wo2,*bo2;
  const float *g1,*be1,*g2,*be2,*g3,*be3;
  const float *W1,*c1,*W2,*c2,*Wr,*br,*Wm,*bm;
  float *mem,*x,*Kc,*Vc,*Q,*Kf,*Vf,*SA,*out;
};

// ---------------- wave32 reductions ----------------
__device__ inline float wsum(float v){
  #pragma unroll
  for (int m = 16; m > 0; m >>= 1) v += __shfl_xor(v, m, 32);
  return v;
}
__device__ inline float wmaxr(float v){
  #pragma unroll
  for (int m = 16; m > 0; m >>= 1) v = fmaxf(v, __shfl_xor(v, m, 32));
  return v;
}

// ---------------- WMMA tile loaders (fp32 source -> f16 regs) ----------------
// A tile 16x32 (MxK), source row-major [M x ld]. CDNA5 16-bit A layout:
// lane l: m = l&15; kbase = (l>>4)*8; elem i<8 -> k0+kbase+i ; i>=8 -> k0+kbase+16+(i-8)
__device__ inline v16h ldA_f32(const float* S, int ld, int m0, int k0, int lane, int mmax){
  int m = m0 + (lane & 15); if (m >= mmax) m = mmax - 1;
  const float* pr = S + m * ld + k0 + ((lane >> 4) << 3);
  v16h a;
  #pragma unroll
  for (int i = 0; i < 8; ++i){ a[i] = (_Float16)pr[i]; a[i+8] = (_Float16)pr[16+i]; }
  return a;
}
// B tile 32x16 (KxN) where B[k][n] = S[(n0+n)*ld + k0+k]  (weights stored [out][in])
// CDNA5 16-bit B layout: lane l: n = l&15; kbase = (l>>4)*8; elem i<8 -> k0+kbase+i ; i>=8 -> k0+kbase+16+(i-8)
__device__ inline v16h ldBt_f32(const float* S, int ld, int n0, int k0, int lane, int nmax){
  int nn = n0 + (lane & 15); if (nn >= nmax) nn = nmax - 1;
  const float* pr = S + nn * ld + k0 + ((lane >> 4) << 3);
  v16h b;
  #pragma unroll
  for (int i = 0; i < 8; ++i){ b[i] = (_Float16)pr[i]; b[i+8] = (_Float16)pr[16+i]; }
  return b;
}
// B tile 32x16 where B[k][n] = S[(k0+k)*ld + n0+n]  (row-major K x N source)
__device__ inline v16h ldBr_f32(const float* S, int ld, int k0, int n0, int lane, int kmax){
  int n = n0 + (lane & 15); int kb = k0 + ((lane >> 4) << 3);
  v16h b;
  #pragma unroll
  for (int i = 0; i < 8; ++i){
    int k1 = kb + i; if (k1 >= kmax) k1 = kmax - 1;   // clamp: padded rows may be NaN, probs==0
    int k2 = kb + 16 + i; if (k2 >= kmax) k2 = kmax - 1;
    b[i] = (_Float16)S[k1 * ld + n]; b[i+8] = (_Float16)S[k2 * ld + n]; }
  return b;
}

// ---------------- single-wave helpers ----------------
__device__ inline void matvec64(const float* W, const float* b, const float* x,
                                float* y, int lane){
  float a0 = b[lane], a1 = b[lane + 32];
  const float* r0 = W + lane * 64;
  const float* r1 = W + (lane + 32) * 64;
  #pragma unroll 4
  for (int j = 0; j < 64; ++j){ float xj = x[j];
    a0 = fmaf(r0[j], xj, a0); a1 = fmaf(r1[j], xj, a1); }
  y[lane] = a0; y[lane + 32] = a1;
}

__device__ inline void ln_pair(float* buf, const float* g, const float* b, int lane){
  float a0 = buf[lane], a1 = buf[lane + 32];
  float m = wsum(a0 + a1) * (1.0f / 64.0f);
  float d0 = a0 - m, d1 = a1 - m;
  float var = wsum(d0 * d0 + d1 * d1) * (1.0f / 64.0f);
  float inv = rsqrtf(var + 1e-5f);
  buf[lane]      = d0 * inv * g[lane]      + b[lane];
  buf[lane + 32] = d1 * inv * g[lane + 32] + b[lane + 32];
}

// shared tail: h1=LN(x+ov), h2=LN(h1+cross), FFN, h3=LN(h2+ffn), r=Wr@h3+br
// on return: hb = h3, hid[0..30] = r
__device__ inline void decoder_tail(const Par& p, const float* xv, const float* ov,
                                    const float* cr, float* hb, float* hid, int lane){
  hb[lane]      = xv[lane]      + ov[lane];
  hb[lane + 32] = xv[lane + 32] + ov[lane + 32];
  __syncthreads();
  ln_pair(hb, p.g1, p.be1, lane); __syncthreads();
  hb[lane] += cr[lane]; hb[lane + 32] += cr[lane + 32]; __syncthreads();
  ln_pair(hb, p.g2, p.be2, lane); __syncthreads();
  #pragma unroll
  for (int j = 0; j < 4; ++j){
    int o = lane + 32 * j;
    float acc = p.c1[o];
    const float* w = p.W1 + o * 64;
    for (int d = 0; d < 64; ++d) acc = fmaf(w[d], hb[d], acc);
    hid[o] = fmaxf(acc, 0.0f);
  }
  __syncthreads();
  #pragma unroll
  for (int jj = 0; jj < 2; ++jj){
    int o = lane + 32 * jj;
    float acc = p.c2[o];
    const float* w = p.W2 + o * 128;
    for (int j = 0; j < 128; ++j) acc = fmaf(w[j], hid[j], acc);
    hb[o] += acc;                // h2 + ffn (own-lane slots only)
  }
  __syncthreads();
  ln_pair(hb, p.g3, p.be3, lane); __syncthreads();   // hb = h3
  if (lane < 31){
    float acc = p.br[lane];
    const float* w = p.Wr + lane * 64;
    for (int d = 0; d < 64; ++d) acc = fmaf(w[d], hb[d], acc);
    hid[lane] = acc;
  }
  __syncthreads();
}

// ================= kernel 1: mem = audio @ W_audio.T + b_audio (WMMA, K=1024) ===========
__global__ __launch_bounds__(32) void mem_gemm(Par p){
  int rt = blockIdx.x, ct = blockIdx.y, lane = threadIdx.x;
  v8f acc = {};
  for (int k0 = 0; k0 < 1024; k0 += 32){
    v16h a = ldA_f32(p.audio,   1024, rt * 16, k0, lane, TT);
    v16h b = ldBt_f32(p.W_audio,1024, ct * 16, k0, lane, DD);
    acc = __builtin_amdgcn_wmma_f32_16x16x32_f16(false, a, false, b, (short)0, acc, false, false);
  }
  int n = ct * 16 + (lane & 15);
  float bias = p.b_audio[n];
  float* dst = p.mem + (rt * 16 + ((lane >> 4) << 3)) * DD + n;   // padded: no guards
  #pragma unroll
  for (int r = 0; r < 8; ++r) dst[r * DD] = acc[r] + bias;
}

// ================= kernel 2: sequential scan, 1 wave, KV-cached =========================
__global__ __launch_bounds__(32) void seq_kernel(Par p){
  __shared__ float emb[TT * DD];                 // 30 KB
  __shared__ float pe[10 * DD], sty[DD], cl[DD];
  __shared__ float xv[DD], qv[DD], kv[DD], vv[DD], sa[DD], ov[DD], hb[DD];
  __shared__ float hid[128], sc[2 * 128];
  int lane = threadIdx.x;

  // positional encoding base (period 10)
  for (int idx = lane; idx < 10 * DD; idx += 32){
    int pos = idx >> 6, d = idx & 63;
    float div = __expf((float)(d & ~1) * (-9.210340371976184f / 64.0f)); // ln(1e4)
    float ang = (float)pos * div;
    pe[idx] = (d & 1) ? cosf(ang) : sinf(ang);
  }
  sty[lane] = p.obj_W[lane * 5];  sty[lane + 32] = p.obj_W[(lane + 32) * 5];
  // cross_last from mem[119]
  xv[lane] = p.mem[119 * DD + lane]; xv[lane + 32] = p.mem[119 * DD + lane + 32];
  __syncthreads();
  matvec64(p.Wv2, p.bv2, xv, kv, lane); __syncthreads();
  matvec64(p.Wo2, p.bo2, kv, cl, lane); __syncthreads();
  emb[lane] = sty[lane]; emb[lane + 32] = sty[lane + 32];
  __syncthreads();

  for (int t = 0; t < TT - 1; ++t){
    const float* per = pe + (t % 10) * DD;
    xv[lane]      = emb[t * DD + lane]      + per[lane];
    xv[lane + 32] = emb[t * DD + lane + 32] + per[lane + 32];
    __syncthreads();
    matvec64(p.Wk, p.bk, xv, kv, lane);
    matvec64(p.Wv, p.bv, xv, vv, lane);
    matvec64(p.Wq, p.bq, xv, qv, lane);
    __syncthreads();
    p.Kc[t * DD + lane] = kv[lane]; p.Kc[t * DD + lane + 32] = kv[lane + 32];
    p.Vc[t * DD + lane] = vv[lane]; p.Vc[t * DD + lane + 32] = vv[lane + 32];
    __threadfence();                 // order same-wave cross-lane KV store->load

    // masked ALiBi attention over cached K (s <= t)
    #pragma unroll
    for (int hh = 0; hh < 2; ++hh){
      float slope = hh ? (1.0f / 256.0f) : (1.0f / 16.0f);
      float e[4]; float mx = NEGV;
      #pragma unroll
      for (int j = 0; j < 4; ++j){
        int s = lane + 32 * j;
        float val = NEGV;
        if (s <= t){
          float acc = 0.0f;
          const float* kr = p.Kc + s * DD + hh * 32;
          for (int d = 0; d < 32; ++d) acc = fmaf(qv[hh * 32 + d], kr[d], acc);
          val = acc * SCALEV - slope * (float)((t - s) / 10);
        }
        e[j] = val; mx = fmaxf(mx, val);
      }
      mx = wmaxr(mx);
      float sum = 0.0f;
      #pragma unroll
      for (int j = 0; j < 4; ++j){
        int s = lane + 32 * j;
        e[j] = (s <= t) ? __expf(e[j] - mx) : 0.0f;
        sum += e[j];
      }
      sum = wsum(sum); float inv = 1.0f / sum;
      #pragma unroll
      for (int j = 0; j < 4; ++j) sc[hh * 128 + lane + 32 * j] = e[j] * inv;
    }
    __syncthreads();
    #pragma unroll
    for (int pi = 0; pi < 2; ++pi){
      int d2 = lane + 32 * pi; int hh = d2 >> 5; int ddim = d2 & 31;
      float acc = 0.0f;
      const float* vbase = p.Vc + hh * 32 + ddim;
      const float* pr = sc + hh * 128;
      for (int s = 0; s <= t; ++s) acc = fmaf(pr[s], vbase[s * DD], acc);
      sa[d2] = acc;
    }
    __syncthreads();
    matvec64(p.Wo, p.bo, sa, ov, lane); __syncthreads();
    decoder_tail(p, xv, ov, cl, hb, hid, lane);
    // new = Wm @ r + bm + style -> emb[t+1]
    #pragma unroll
    for (int jj = 0; jj < 2; ++jj){
      int o = lane + 32 * jj;
      float acc = p.bm[o] + sty[o];
      const float* w = p.Wm + o * 31;
      for (int j = 0; j < 31; ++j) acc = fmaf(w[j], hid[j], acc);
      emb[(t + 1) * DD + o] = acc;
    }
    __syncthreads();
  }
  // final x = emb + pe  -> workspace (consumed by WMMA kernels)
  for (int idx = lane; idx < TT * DD; idx += 32){
    int pos = idx >> 6, d = idx & 63;
    p.x[idx] = emb[idx] + pe[(pos % 10) * DD + d];
  }
}

// ================= kernel 3: Q/K/V = x @ W.T + b (WMMA) =================================
__global__ __launch_bounds__(32) void qkv_gemm(Par p){
  int rt = blockIdx.x, ct = blockIdx.y, which = blockIdx.z, lane = threadIdx.x;
  const float* W = (which == 0) ? p.Wq : (which == 1) ? p.Wk : p.Wv;
  const float* b = (which == 0) ? p.bq : (which == 1) ? p.bk : p.bv;
  float*       O = (which == 0) ? p.Q  : (which == 1) ? p.Kf : p.Vf;
  v8f acc = {};
  #pragma unroll
  for (int k0 = 0; k0 < 64; k0 += 32){
    v16h a  = ldA_f32(p.x, 64, rt * 16, k0, lane, TT);
    v16h bb = ldBt_f32(W, 64, ct * 16, k0, lane, DD);
    acc = __builtin_amdgcn_wmma_f32_16x16x32_f16(false, a, false, bb, (short)0, acc, false, false);
  }
  int n = ct * 16 + (lane & 15);
  float bias = b[n];
  float* dst = O + (rt * 16 + ((lane >> 4) << 3)) * DD + n;       // padded: no guards
  #pragma unroll
  for (int r = 0; r < 8; ++r) dst[r * DD] = acc[r] + bias;
}

// ================= kernel 4: attention (scores WMMA, softmax, A@V WMMA) =================
__global__ __launch_bounds__(32) void attn_kernel(Par p){
  __shared__ float sc[16 * 128];
  int rt = blockIdx.x, hh = blockIdx.y, lane = threadIdx.x;
  float slope = hh ? (1.0f / 256.0f) : (1.0f / 16.0f);
  int lr = lane & 15, hf = lane >> 4;

  v16h aq = ldA_f32(p.Q, 64, rt * 16, hh * 32, lane, TT);
  #pragma unroll
  for (int ct = 0; ct < 8; ++ct){
    v16h bk = ldBt_f32(p.Kf, 64, ct * 16, hh * 32, lane, TT);
    v8f acc = {};
    acc = __builtin_amdgcn_wmma_f32_16x16x32_f16(false, aq, false, bk, (short)0, acc, false, false);
    int n = ct * 16 + lr;
    #pragma unroll
    for (int r = 0; r < 8; ++r){
      int ml = r + 8 * hf;
      int m = rt * 16 + ml;
      float val = NEGV;
      if (m < TT && n <= m) val = acc[r] * SCALEV - slope * (float)((m - n) / 10);
      sc[ml * 128 + n] = val;
    }
  }
  __syncthreads();
  // row-wise softmax over 128 (padded) columns
  for (int r = 0; r < 16; ++r){
    float v0 = sc[r*128+lane], v1 = sc[r*128+lane+32], v2 = sc[r*128+lane+64], v3 = sc[r*128+lane+96];
    float mx = wmaxr(fmaxf(fmaxf(v0, v1), fmaxf(v2, v3)));
    v0 = __expf(v0 - mx); v1 = __expf(v1 - mx); v2 = __expf(v2 - mx); v3 = __expf(v3 - mx);
    float inv = 1.0f / wsum(v0 + v1 + v2 + v3);
    sc[r*128+lane] = v0*inv; sc[r*128+lane+32] = v1*inv;
    sc[r*128+lane+64] = v2*inv; sc[r*128+lane+96] = v3*inv;
  }
  __syncthreads();
  // SA = probs @ V  (K padded 120->128 with zero probs; V rows clamped)
  #pragma unroll
  for (int ctv = 0; ctv < 2; ++ctv){
    v8f acc = {};
    #pragma unroll
    for (int kc = 0; kc < 4; ++kc){
      v16h ap = ldA_f32(sc, 128, 0, kc * 32, lane, 16);
      v16h bv = ldBr_f32(p.Vf, 64, kc * 32, hh * 32 + ctv * 16, lane, TT);
      acc = __builtin_amdgcn_wmma_f32_16x16x32_f16(false, ap, false, bv, (short)0, acc, false, false);
    }
    int n = hh * 32 + ctv * 16 + lr;
    float* dst = p.SA + (rt * 16 + 8 * hf) * DD + n;              // padded: no guards
    #pragma unroll
    for (int r = 0; r < 8; ++r) dst[r * DD] = acc[r];
  }
}

// ================= kernel 5: per-row tail (out-proj, LNs, cross, FFN, Wr) ===============
__global__ __launch_bounds__(32) void final_rows(Par p){
  __shared__ float xv[DD], sv[DD], t1[DD], crs[DD], hb[DD], hid[128];
  int i = blockIdx.x, lane = threadIdx.x;
  xv[lane] = p.x[i*DD+lane];  xv[lane+32] = p.x[i*DD+lane+32];
  sv[lane] = p.SA[i*DD+lane]; sv[lane+32] = p.SA[i*DD+lane+32];
  __syncthreads();
  matvec64(p.Wo, p.bo, sv, t1, lane); __syncthreads();   // t1 = Wo@sa + bo
  // per-row cross = Wo2 @ (Wv2 @ mem_i + bv2) + bo2
  sv[lane] = p.mem[i*DD+lane]; sv[lane+32] = p.mem[i*DD+lane+32];
  __syncthreads();
  matvec64(p.Wv2, p.bv2, sv, crs, lane); __syncthreads();
  matvec64(p.Wo2, p.bo2, crs, sv, lane); __syncthreads(); // sv = cross
  decoder_tail(p, xv, t1, sv, hb, hid, lane);
  if (lane < 31) p.out[i * 31 + lane] = hid[lane];
}

extern "C" void kernel_launch(void* const* d_in, const int* in_sizes, int n_in,
                              void* d_out, int out_size, void* d_ws, size_t ws_size,
                              hipStream_t stream) {
  (void)in_sizes; (void)n_in; (void)out_size; (void)ws_size;
  Par p;
  p.audio  =(const float*)d_in[0];  p.W_audio=(const float*)d_in[1];
  p.b_audio=(const float*)d_in[2];  p.obj_W  =(const float*)d_in[3];
  p.Wq =(const float*)d_in[4];  p.bq =(const float*)d_in[5];
  p.Wk =(const float*)d_in[6];  p.bk =(const float*)d_in[7];
  p.Wv =(const float*)d_in[8];  p.bv =(const float*)d_in[9];
  p.Wo =(const float*)d_in[10]; p.bo =(const float*)d_in[11];
  p.Wv2=(const float*)d_in[12]; p.bv2=(const float*)d_in[13];
  p.Wo2=(const float*)d_in[14]; p.bo2=(const float*)d_in[15];
  p.g1 =(const float*)d_in[16]; p.be1=(const float*)d_in[17];
  p.g2 =(const float*)d_in[18]; p.be2=(const float*)d_in[19];
  p.g3 =(const float*)d_in[20]; p.be3=(const float*)d_in[21];
  p.W1 =(const float*)d_in[22]; p.c1 =(const float*)d_in[23];
  p.W2 =(const float*)d_in[24]; p.c2 =(const float*)d_in[25];
  p.Wr =(const float*)d_in[26]; p.br =(const float*)d_in[27];
  p.Wm =(const float*)d_in[28]; p.bm =(const float*)d_in[29];

  float* w = (float*)d_ws;                 // all regions padded to TP*DD = 8192 floats
  p.mem = w;             p.x  = w + 8192;
  p.Kc  = w + 16384;     p.Vc = w + 24576;
  p.Q   = w + 32768;     p.Kf = w + 40960;  p.Vf = w + 49152;
  p.SA  = w + 57344;
  p.out = (float*)d_out;

  mem_gemm  <<<dim3(8, 4),    32, 0, stream>>>(p);
  seq_kernel<<<1,             32, 0, stream>>>(p);
  qkv_gemm  <<<dim3(8, 4, 3), 32, 0, stream>>>(p);
  attn_kernel<<<dim3(8, 2),   32, 0, stream>>>(p);
  final_rows<<<120,           32, 0, stream>>>(p);
}
